// MSARowAttentionWithPairBias_22479858827761
// MI455X (gfx1250) — matrix operand order repaired and checked by reading, not compiled, not run on hardware
//
#include <hip/hip_runtime.h>
#include <hip/hip_bf16.h>
#include <math.h>

typedef __attribute__((ext_vector_type(16))) _Float16 v16h;
typedef __attribute__((ext_vector_type(8)))  _Float16 v8h;
typedef __attribute__((ext_vector_type(8)))  float    v8f;

#define S_DIM   8
#define N_DIM   256
#define MSA_D   256
#define PAIR_D  128
#define H_DIM   8
#define C_DIM   256      // D*H
#define QKV_N   768
#define EPSLN   1e-5f

__device__ __forceinline__ float wave_reduce_add(float v) {
  #pragma unroll
  for (int m = 16; m >= 1; m >>= 1) v += __shfl_xor(v, m, 32);
  return v;
}

// ---------------------------------------------------------------------------
// Convert f32 weight [K x N] row-major -> f16 transposed [N x K] row-major
// ---------------------------------------------------------------------------
__global__ void cvt_transpose_f16(const float* __restrict__ src, _Float16* __restrict__ dst,
                                  int K, int N) {
  int idx = blockIdx.x * blockDim.x + threadIdx.x;
  if (idx < K * N) {
    int k = idx / N, n = idx % N;
    dst[(size_t)n * K + k] = (_Float16)src[idx];
  }
}

// ---------------------------------------------------------------------------
// LayerNorm over last dim (256) of msa rows -> f16.  One wave per row.
// ---------------------------------------------------------------------------
__global__ void ln_msa_to_f16(const float* __restrict__ msa, const float* __restrict__ w,
                              const float* __restrict__ b, _Float16* __restrict__ xo) {
  int wid = threadIdx.x >> 5, lane = threadIdx.x & 31;
  int row = blockIdx.x * 8 + wid;                 // 2048 rows total
  const float* x = msa + (size_t)row * MSA_D;
  float v[8], s = 0.f;
  #pragma unroll
  for (int t = 0; t < 8; ++t) { v[t] = x[lane + 32 * t]; s += v[t]; }
  s = wave_reduce_add(s);
  float mu = s * (1.f / MSA_D);
  float vs = 0.f;
  #pragma unroll
  for (int t = 0; t < 8; ++t) { float d = v[t] - mu; vs += d * d; }
  vs = wave_reduce_add(vs);
  float r = rsqrtf(vs * (1.f / MSA_D) + EPSLN);
  #pragma unroll
  for (int t = 0; t < 8; ++t) {
    int k = lane + 32 * t;
    xo[(size_t)row * MSA_D + k] = (_Float16)((v[t] - mu) * r * w[k] + b[k]);
  }
}

// ---------------------------------------------------------------------------
// Pair bias: LN(pair[i,j,:128]) @ w_b[128,8] -> bias[i,j,8].  One wave / (i,j).
// ---------------------------------------------------------------------------
__global__ void ln_pair_bias(const float* __restrict__ pair, const float* __restrict__ w,
                             const float* __restrict__ b, const float* __restrict__ wb,
                             float* __restrict__ bias) {
  int wid = threadIdx.x >> 5, lane = threadIdx.x & 31;
  size_t ij = (size_t)blockIdx.x * 8 + wid;       // 65536 pairs total
  const float* x = pair + ij * PAIR_D;
  float v[4], s = 0.f;
  #pragma unroll
  for (int t = 0; t < 4; ++t) { v[t] = x[lane + 32 * t]; s += v[t]; }
  s = wave_reduce_add(s);
  float mu = s * (1.f / PAIR_D);
  float vs = 0.f;
  #pragma unroll
  for (int t = 0; t < 4; ++t) { float d = v[t] - mu; vs += d * d; }
  vs = wave_reduce_add(vs);
  float r = rsqrtf(vs * (1.f / PAIR_D) + EPSLN);
  float acc[H_DIM];
  #pragma unroll
  for (int h = 0; h < H_DIM; ++h) acc[h] = 0.f;
  #pragma unroll
  for (int t = 0; t < 4; ++t) {
    int k = lane + 32 * t;
    float xn = (v[t] - mu) * r * w[k] + b[k];
    #pragma unroll
    for (int h = 0; h < H_DIM; ++h) acc[h] += xn * wb[k * H_DIM + h];
  }
  float myout = 0.f;
  #pragma unroll
  for (int h = 0; h < H_DIM; ++h) {
    float t = wave_reduce_add(acc[h]);
    if (lane == h) myout = t;
  }
  if (lane < H_DIM) bias[ij * H_DIM + lane] = myout;
}

// ---------------------------------------------------------------------------
// WMMA GEMM: C[MxN] (f32, optional row-bias) = A[MxK] (f16) * Bt[NxK]^T (f16)
// One wave -> one 16x16 tile; K-loop step 32 via v_wmma_f32_16x16x32_f16.
// ---------------------------------------------------------------------------
__global__ void gemm_wmma_f16(const _Float16* __restrict__ A, const _Float16* __restrict__ Bt,
                              float* __restrict__ C, const float* __restrict__ bias,
                              int M, int N, int K) {
  int wid = threadIdx.x >> 5, lane = threadIdx.x & 31;
  int tile   = blockIdx.x * 8 + wid;
  int tilesN = N >> 4;
  int tm = tile / tilesN, tn = tile % tilesN;
  int mr  = lane & 15;
  int kbA = (lane >> 4) << 3;                     // A frag: lanes 16-31 start at K=8
  int kbB = (lane >> 4) << 3;                     // B frag: lanes 16-31 start at K=8
  const _Float16* ap = A  + ((size_t)(tm * 16 + mr)) * K + kbA;
  const _Float16* bp = Bt + ((size_t)(tn * 16 + mr)) * K + kbB;
  v8f c = {};
  for (int k0 = 0; k0 < K; k0 += 32) {
    v8h al = *(const v8h*)(ap + k0);
    v8h ah = *(const v8h*)(ap + k0 + 16);
    v8h bl = *(const v8h*)(bp + k0);
    v8h bh = *(const v8h*)(bp + k0 + 16);
    v16h a = __builtin_shufflevector(al, ah, 0,1,2,3,4,5,6,7,8,9,10,11,12,13,14,15);
    v16h bb = __builtin_shufflevector(bl, bh, 0,1,2,3,4,5,6,7,8,9,10,11,12,13,14,15);
    c = __builtin_amdgcn_wmma_f32_16x16x32_f16(false, a, false, bb,
                                               (short)0, c, false, false);
  }
  int cn    = tn * 16 + (lane & 15);
  int mbase = tm * 16 + ((lane >> 4) << 3);
  float badd = bias ? bias[cn] : 0.f;
  #pragma unroll
  for (int r = 0; r < 8; ++r)
    C[(size_t)(mbase + r) * N + cn] = c[r] + badd;
}

// ---------------------------------------------------------------------------
// Per-channel attention: for each (s,i,c): softmax_j(q*k_j*scale + b[i,j,h]),
// weighted sum of v, gate with sigmoid(x@w_g), emit f16 for the output GEMM.
// Block = one (s,i) row, 256 threads = channels; bias tile staged in LDS.
// ---------------------------------------------------------------------------
__global__ void attn_softmax_kernel(const float* __restrict__ qkv, const float* __restrict__ pairb,
                                    const float* __restrict__ gpre, _Float16* __restrict__ aout) {
  __shared__ float bsh[N_DIM * H_DIM];            // 8 KB: b[i, j=0..255, h=0..7]
  int s = blockIdx.x / N_DIM;
  int i = blockIdx.x % N_DIM;
  int c = threadIdx.x;
  const float* brow = pairb + (size_t)i * N_DIM * H_DIM;
  #pragma unroll
  for (int t = 0; t < 8; ++t) bsh[c + 256 * t] = brow[c + 256 * t];
  __syncthreads();

  int h = c & 7;
  size_t rowq = ((size_t)s * N_DIM + i) * QKV_N;
  float q = qkv[rowq + c] * 0.17677669529663687f;       // D^-0.5, D=32
  const float* kp = qkv + (size_t)s * N_DIM * QKV_N + 256 + c;
  const float* vp = qkv + (size_t)s * N_DIM * QKV_N + 512 + c;

  float mx = -INFINITY, l = 0.f, acc = 0.f;
  for (int j = 0; j < N_DIM; ++j) {
    float kj = kp[(size_t)j * QKV_N];
    float vj = vp[(size_t)j * QKV_N];
    float sc = fmaf(q, kj, bsh[j * H_DIM + h]);
    float nm = fmaxf(mx, sc);
    float corr = __expf(mx - nm);
    float p    = __expf(sc - nm);
    l   = l   * corr + p;
    acc = acc * corr + p * vj;
    mx = nm;
  }
  float g = 1.f / (1.f + __expf(-gpre[((size_t)s * N_DIM + i) * C_DIM + c]));
  aout[((size_t)s * N_DIM + i) * C_DIM + c] = (_Float16)(g * acc / l);
}

// ---------------------------------------------------------------------------
extern "C" void kernel_launch(void* const* d_in, const int* in_sizes, int n_in,
                              void* d_out, int out_size, void* d_ws, size_t ws_size,
                              hipStream_t stream) {
  (void)in_sizes; (void)n_in; (void)out_size; (void)ws_size;
  const float* msa     = (const float*)d_in[0];
  const float* pair    = (const float*)d_in[1];
  const float* ln_w    = (const float*)d_in[2];
  const float* ln_b    = (const float*)d_in[3];
  const float* w_qkv   = (const float*)d_in[4];
  const float* lnp_w   = (const float*)d_in[5];
  const float* lnp_b   = (const float*)d_in[6];
  const float* w_b     = (const float*)d_in[7];
  const float* w_g     = (const float*)d_in[8];
  const float* w_out   = (const float*)d_in[9];
  const float* b_out   = (const float*)d_in[10];
  float* out = (float*)d_out;

  const int ROWS = S_DIM * N_DIM;                 // 2048
  // --- workspace carve-up (~13.3 MB total) ---
  _Float16* x16   = (_Float16*)d_ws;              // 2048*256
  _Float16* wqkvT = x16   + (size_t)ROWS * MSA_D; // 768*256
  _Float16* wgT   = wqkvT + (size_t)QKV_N * MSA_D;// 256*256
  _Float16* woutT = wgT   + (size_t)C_DIM * MSA_D;// 256*256
  _Float16* a16   = woutT + (size_t)MSA_D * C_DIM;// 2048*256
  float* qkvf  = (float*)(a16 + (size_t)ROWS * C_DIM);   // 2048*768
  float* gpre  = qkvf  + (size_t)ROWS * QKV_N;           // 2048*256
  float* pairb = gpre  + (size_t)ROWS * C_DIM;           // 256*256*8

  // 1) weight conversions (f32 -> f16 transposed)
  cvt_transpose_f16<<<(MSA_D * QKV_N + 255) / 256, 256, 0, stream>>>(w_qkv, wqkvT, MSA_D, QKV_N);
  cvt_transpose_f16<<<(MSA_D * C_DIM + 255) / 256, 256, 0, stream>>>(w_g, wgT, MSA_D, C_DIM);
  cvt_transpose_f16<<<(C_DIM * MSA_D + 255) / 256, 256, 0, stream>>>(w_out, woutT, C_DIM, MSA_D);

  // 2) layernorms
  ln_msa_to_f16<<<ROWS / 8, 256, 0, stream>>>(msa, ln_w, ln_b, x16);
  ln_pair_bias<<<(N_DIM * N_DIM) / 8, 256, 0, stream>>>(pair, lnp_w, lnp_b, w_b, pairb);

  // 3) qkv = x @ w_qkv   [2048x768], gate-pre = x @ w_g  [2048x256]
  gemm_wmma_f16<<<(ROWS / 16) * (QKV_N / 16) / 8, 256, 0, stream>>>(x16, wqkvT, qkvf, nullptr,
                                                                    ROWS, QKV_N, MSA_D);
  gemm_wmma_f16<<<(ROWS / 16) * (C_DIM / 16) / 8, 256, 0, stream>>>(x16, wgT, gpre, nullptr,
                                                                    ROWS, C_DIM, MSA_D);

  // 4) fused per-channel softmax + gating
  attn_softmax_kernel<<<ROWS, 256, 0, stream>>>(qkvf, pairb, gpre, a16);

  // 5) output projection + bias -> d_out
  gemm_wmma_f16<<<(ROWS / 16) * (MSA_D / 16) / 8, 256, 0, stream>>>(a16, woutT, out, b_out,
                                                                    ROWS, MSA_D, C_DIM);
}
